// EmbeddingBlock_5119601016933
// MI455X (gfx1250) — matrix operand order repaired and checked by reading, not compiled, not run on hardware
//
#include <hip/hip_runtime.h>
#include <hip/hip_bf16.h>
#include <cstdint>

#define HIDDEN 128
#define NRAD   16
#define ROWW   136   // 128 + 8 bf16 pad -> row stride 272B == 68 DW == 4 mod 64 banks

typedef __attribute__((ext_vector_type(16))) __bf16 v16bf;
typedef __attribute__((ext_vector_type(8)))  float  v8f;
typedef unsigned int u32x4 __attribute__((ext_vector_type(4)));
typedef int          i32x4 __attribute__((ext_vector_type(4)));
typedef int          i32x8 __attribute__((ext_vector_type(8)));

union Frag16 {
  v16bf  v;
  __bf16 e[16];
  uint4  q[2];
};

union Pack4 {
  uint2  u;
  __bf16 e[4];
};

__device__ __forceinline__ float silu_f(float x) {
  // x * rcp(1+exp(-x)) : v_exp_f32 + v_rcp_f32, no IEEE div expansion
  return x * __builtin_amdgcn_rcpf(1.0f + __expf(-x));
}

__device__ __forceinline__ v16bf ldB32(const __bf16* p) {
  Frag16 f;
  const uint4* q = (const uint4*)p;
  f.q[0] = q[0];
  f.q[1] = q[1];
  return f.v;
}

__device__ __forceinline__ v16bf ldA_lds(const __bf16* lo, const __bf16* hi) {
  Frag16 f;
  f.q[0] = *(const uint4*)(lo);
  f.q[1] = *(const uint4*)(hi);
  return f.v;
}

// LDS byte offset of a __shared__ pointer: low 32 bits of the flat address
__device__ __forceinline__ unsigned lds_off(const void* p) {
  return (unsigned)(unsigned long long)(uintptr_t)p;
}

// TDM gather: 16 rows (16-bit indices) of a [nRows][128] bf16 tensor -> LDS,
// 16B pad inserted after every 64 DWORDs (256B row) => 272B LDS row stride.
__device__ __forceinline__ void tdm_gather16_bf16(const __bf16* src, int nRows,
                                                  unsigned lds_byte_off,
                                                  const unsigned* idx8) {
  unsigned long long ga = (unsigned long long)(uintptr_t)src;
  u32x4 g0;
  g0[0] = 1u | (1u << 31);                 // count=1, 16-bit indices, gather_mode=1
  g0[1] = lds_byte_off;                    // LDS destination
  g0[2] = (unsigned)ga;                    // global_addr[31:0]
  g0[3] = (unsigned)(ga >> 32) | (2u << 30); // global_addr[56:32] | type=2
  i32x8 g1;
  g1[0] = (1 << 16) | (1 << 20) | (5 << 22) | (3 << 25); // 2B data, pad_en, 64DW interval, 4DW pad
  g1[1] = (HIDDEN << 16);                  // tensor_dim0 = 128 (bits 79:48 low part)
  g1[2] = (int)(((unsigned)nRows & 0xffffu) << 16); // tensor_dim1 low16 @bit80
  g1[3] = (nRows >> 16) | (HIDDEN << 16);  // tensor_dim1 hi | tile_dim0 = 128
  g1[4] = 16;                              // tile_dim1 = 16 gather indices
  g1[5] = HIDDEN;                          // tensor_dim0_stride = 128
  g1[6] = 0;
  g1[7] = 0;
  i32x4 g2 = { (int)idx8[0], (int)idx8[1], (int)idx8[2], (int)idx8[3] };
  i32x4 g3 = { (int)idx8[4], (int)idx8[5], (int)idx8[6], (int)idx8[7] };
#if defined(__clang_major__) && (__clang_major__ >= 23)
  i32x8 gz = {0, 0, 0, 0, 0, 0, 0, 0};
  __builtin_amdgcn_tensor_load_to_lds(g0, g1, g2, g3, gz, 0);
#else
  __builtin_amdgcn_tensor_load_to_lds(g0, g1, g2, g3, 0);
#endif
}

__global__ void eb_prep_weights(const float* __restrict__ W_rbf,
                                const float* __restrict__ W_lin,
                                __bf16* __restrict__ wr,
                                __bf16* __restrict__ wl) {
  int t = blockIdx.x * blockDim.x + threadIdx.x;
  if (t < HIDDEN * 32) {
    int n = t >> 5, k = t & 31;
    wr[t] = (k < NRAD) ? (__bf16)W_rbf[n * NRAD + k] : (__bf16)0.0f;
  }
  if (t < HIDDEN * 3 * HIDDEN) {
    wl[t] = (__bf16)W_lin[t];
  }
}

// h[node] = (bf16) emb[x[node]]   (4 elems per thread)
__global__ void eb_prep_h(const int* __restrict__ x,
                          const float* __restrict__ emb,
                          __bf16* __restrict__ hb, int nN) {
  int t = blockIdx.x * blockDim.x + threadIdx.x;
  int node = t >> 5;
  if (node >= nN) return;
  int c = (t & 31) << 2;
  float4 v = *(const float4*)(emb + (size_t)x[node] * HIDDEN + c);
  Pack4 p;
  p.e[0] = (__bf16)v.x; p.e[1] = (__bf16)v.y;
  p.e[2] = (__bf16)v.z; p.e[3] = (__bf16)v.w;
  *(uint2*)(hb + (size_t)node * HIDDEN + c) = p.u;
}

__global__ __launch_bounds__(128)
void eb_edge_kernel(const int*   __restrict__ xtype,
                    const float* __restrict__ rbf,
                    const int*   __restrict__ ei,
                    const int*   __restrict__ ej,
                    const float* __restrict__ emb,
                    const float* __restrict__ b_rbf,
                    const float* __restrict__ b_lin,
                    const __bf16* __restrict__ wr,   // [128][32]
                    const __bf16* __restrict__ wl,   // [128][384]
                    const __bf16* __restrict__ hbf,  // [nN][128] (TDM path)
                    float* __restrict__ out,
                    int nE, int nN, int use_tdm) {
  // 64 edges per block; K-sections split into three bank-rotated arrays
  __shared__ __align__(16) __bf16 hIb[64][ROWW];
  __shared__ __align__(16) __bf16 hJb[64][ROWW];
  __shared__ __align__(16) __bf16 rBb[64][ROWW];

  const int tid  = threadIdx.x;
  const int lane = tid & 31;
  const int w    = tid >> 5;         // wave 0..3 (owns N-tiles 2w, 2w+1)
  const int lm   = lane & 15;
  const int hf   = lane >> 4;
  const int e0   = blockIdx.x << 6;  // 64 edges per block
  const int ew0  = e0 + (w << 4);    // this wave's 16 edges (m-tile w)

  __builtin_prefetch(wl, 0, 1);

  int eIdx = ew0 + lm;
  if (eIdx >= nE) eIdx = nE - 1;
  int iv = ei[eIdx];
  int jv = ej[eIdx];

  // ---- endpoint feature gather into hIb/hJb rows [w*16 .. w*16+15] ----
  if (use_tdm) {
    unsigned gi[8], gj[8];
#pragma unroll
    for (int k2 = 0; k2 < 8; ++k2) {
      unsigned a0 = (unsigned)__shfl(iv, 2 * k2, 32) & 0xffffu;
      unsigned a1 = (unsigned)__shfl(iv, 2 * k2 + 1, 32) & 0xffffu;
      unsigned b0 = (unsigned)__shfl(jv, 2 * k2, 32) & 0xffffu;
      unsigned b1 = (unsigned)__shfl(jv, 2 * k2 + 1, 32) & 0xffffu;
      gi[k2] = a0 | (a1 << 16);
      gj[k2] = b0 | (b1 << 16);
    }
    tdm_gather16_bf16(hbf, nN, lds_off(&hIb[w << 4][0]), gi);
    tdm_gather16_bf16(hbf, nN, lds_off(&hJb[w << 4][0]), gj);
  } else {
    int ti = xtype[iv];
    int tj = xtype[jv];
    for (int r = 0; r < 16; ++r) {
      int t_i = __shfl(ti, r, 32);
      int t_j = __shfl(tj, r, 32);
      float4 a4 = *(const float4*)(emb + (size_t)t_i * HIDDEN + (lane << 2));
      float4 b4 = *(const float4*)(emb + (size_t)t_j * HIDDEN + (lane << 2));
      Pack4 pa, pb;
      pa.e[0] = (__bf16)a4.x; pa.e[1] = (__bf16)a4.y;
      pa.e[2] = (__bf16)a4.z; pa.e[3] = (__bf16)a4.w;
      pb.e[0] = (__bf16)b4.x; pb.e[1] = (__bf16)b4.y;
      pb.e[2] = (__bf16)b4.z; pb.e[3] = (__bf16)b4.w;
      *(uint2*)&hIb[(w << 4) + r][lane << 2] = pa.u;
      *(uint2*)&hJb[(w << 4) + r][lane << 2] = pb.u;
    }
  }

  // ---- rbf projection for this wave's 16 edges: A = 16x32 (K zero-padded) ----
  Frag16 af;
  {
    const float* rp = rbf + (size_t)eIdx * NRAD + (hf << 3);
    float4 r0 = *(const float4*)rp;
    float4 r1 = *(const float4*)(rp + 4);
    af.e[0] = (__bf16)r0.x; af.e[1] = (__bf16)r0.y;
    af.e[2] = (__bf16)r0.z; af.e[3] = (__bf16)r0.w;
    af.e[4] = (__bf16)r1.x; af.e[5] = (__bf16)r1.y;
    af.e[6] = (__bf16)r1.z; af.e[7] = (__bf16)r1.w;
#pragma unroll
    for (int k = 8; k < 16; ++k) af.e[k] = (__bf16)0.0f;
  }

  const v8f vzero = {0.f, 0.f, 0.f, 0.f, 0.f, 0.f, 0.f, 0.f};

#pragma unroll
  for (int n = 0; n < 8; ++n) {
    v16bf b = ldA_lds(wr + ((n * 16 + lm) << 5) + (hf << 3), wr + ((n * 16 + lm) << 5) + 16 + (hf << 3));
    v8f c = __builtin_amdgcn_wmma_f32_16x16x32_bf16(
        false, af.v, false, b, (short)0, vzero, false, false);
    int col = n * 16 + lm;
    float bb = b_rbf[col];
#pragma unroll
    for (int v = 0; v < 8; ++v) {
      int m = (w << 4) + v + (hf << 3);
      rBb[m][col] = (__bf16)silu_f(c[v] + bb);
    }
  }

  if (use_tdm) __builtin_amdgcn_s_wait_tensorcnt(0);
  __syncthreads();

  // ---- main GEMM: [64 x 384] x [384 x 128]; wave w owns N-tiles 2w,2w+1 ----
  v8f acc[2][4];
#pragma unroll
  for (int nn = 0; nn < 2; ++nn)
#pragma unroll
    for (int mt = 0; mt < 4; ++mt) acc[nn][mt] = vzero;

#pragma unroll
  for (int kt = 0; kt < 12; ++kt) {
    const int k0 = kt << 5;
    const __bf16* ab;
    int kl;
    if (kt < 4)      { ab = &hIb[0][0]; kl = k0; }
    else if (kt < 8) { ab = &hJb[0][0]; kl = k0 - HIDDEN; }
    else             { ab = &rBb[0][0]; kl = k0 - 2 * HIDDEN; }

    v16bf a[4];
#pragma unroll
    for (int mt = 0; mt < 4; ++mt) {
      const __bf16* rowp = ab + (size_t)((mt << 4) + lm) * ROWW + kl;
      a[mt] = ldA_lds(rowp + (hf << 3), rowp + 16 + (hf << 3));
    }
#pragma unroll
    for (int nn = 0; nn < 2; ++nn) {
      const __bf16* bp =
          wl + (size_t)(((w * 2 + nn) * 16 + lm)) * (3 * HIDDEN) + k0 + (hf << 3);
      v16bf b = ldA_lds(bp, bp + 16);
#pragma unroll
      for (int mt = 0; mt < 4; ++mt)
        acc[nn][mt] = __builtin_amdgcn_wmma_f32_16x16x32_bf16(
            false, a[mt], false, b, (short)0, acc[nn][mt], false, false);
    }
  }

  // ---- epilogue: bias + silu, f32 stores ----
#pragma unroll
  for (int nn = 0; nn < 2; ++nn) {
    int col = (w * 2 + nn) * 16 + lm;
    float bb = b_lin[col];
#pragma unroll
    for (int mt = 0; mt < 4; ++mt) {
#pragma unroll
      for (int v = 0; v < 8; ++v) {
        int e = e0 + (mt << 4) + v + (hf << 3);
        if (e < nE) out[(size_t)e * HIDDEN + col] = silu_f(acc[nn][mt][v] + bb);
      }
    }
  }
}

extern "C" void kernel_launch(void* const* d_in, const int* in_sizes, int n_in,
                              void* d_out, int out_size, void* d_ws, size_t ws_size,
                              hipStream_t stream) {
  const int*   x     = (const int*)d_in[0];
  const float* rbf   = (const float*)d_in[1];
  const int*   ei    = (const int*)d_in[2];
  const int*   ej    = (const int*)d_in[3];
  const float* emb   = (const float*)d_in[4];
  const float* W_rbf = (const float*)d_in[5];
  const float* b_rbf = (const float*)d_in[6];
  const float* W_lin = (const float*)d_in[7];
  const float* b_lin = (const float*)d_in[8];
  float* out = (float*)d_out;

  const int nN = in_sizes[0];  // node count from x[]
  const int nE = in_sizes[2];  // edge count from i[]

  // workspace: wr [128*32] | wl [128*384] | h_bf16 [nN*128]
  __bf16* wr = (__bf16*)d_ws;
  __bf16* wl = wr + HIDDEN * 32;
  __bf16* hb = wl + HIDDEN * 3 * HIDDEN;
  size_t need = sizeof(__bf16) *
                ((size_t)HIDDEN * 32 + (size_t)HIDDEN * 3 * HIDDEN + (size_t)nN * HIDDEN);
  // TDM gather path needs the bf16 node-feature table + 16-bit row indices
  const int use_tdm = (ws_size >= need && nN <= 65536) ? 1 : 0;

  const int prep_threads = HIDDEN * 3 * HIDDEN;
  eb_prep_weights<<<(prep_threads + 255) / 256, 256, 0, stream>>>(W_rbf, W_lin, wr, wl);
  if (use_tdm) {
    eb_prep_h<<<(nN * 32 + 255) / 256, 256, 0, stream>>>(x, emb, hb, nN);
  }

  const int blocks = (nE + 63) / 64;
  eb_edge_kernel<<<blocks, 128, 0, stream>>>(x, rbf, ei, ej, emb, b_rbf, b_lin,
                                             wr, wl, hb, out, nE, nN, use_tdm);
}
